// GATv2Encoder_52845277610396
// MI455X (gfx1250) — compile-verified, matrix operand order checked by reading, not run on hardware
//
#include <hip/hip_runtime.h>
#include <hip/hip_fp16.h>

typedef __attribute__((ext_vector_type(16))) _Float16 v16h;
typedef __attribute__((ext_vector_type(8)))  float    v8f;

#define NEG_SLOPE 0.2f

// Order-preserving float -> u32 mapping for atomicMax on signed floats.
__device__ __forceinline__ unsigned fmap(float f) {
  int i = __float_as_int(f);
  return (i >= 0) ? ((unsigned)i | 0x80000000u) : ~((unsigned)i);
}
__device__ __forceinline__ float funmap(unsigned u) {
  int i = (u & 0x80000000u) ? (int)(u & 0x7FFFFFFFu) : (int)~u;
  return __int_as_float(i);
}

__global__ void fill_u32_kernel(unsigned* __restrict__ p, unsigned val, long long n) {
  long long i = (long long)blockIdx.x * blockDim.x + threadIdx.x;
  if (i < n) p[i] = val;
}

__global__ void f32_to_f16_kernel(const float* __restrict__ in, _Float16* __restrict__ out, int n) {
  int i = blockIdx.x * blockDim.x + threadIdx.x;
  if (i < n) out[i] = (_Float16)in[i];
}

// Convert + swizzle a 128x128 f32 weight matrix into WMMA B-fragment order:
// out[(((kb*8 + ct)*32 + lane)*16 + h] = W[k, n] where the (lane,h)->(k,n)
// mapping matches the CDNA5 16-bit B 32x16 VGPR layout (ISA 7.12.2).
// The GEMM then reads each lane's 16 halves as one contiguous 32B load.
__global__ void wswz_kernel(const float* __restrict__ W, _Float16* __restrict__ out) {
  int t = blockIdx.x * blockDim.x + threadIdx.x;   // 0 .. 16383
  if (t >= 128 * 128) return;
  int h    = t & 15;
  int lane = (t >> 4) & 31;
  int frag = t >> 9;          // 0..31
  int ct   = frag & 7;        // column tile
  int kb   = frag >> 3;       // k block (32 wide)
  int lh   = lane & 15;
  int koff = (lane < 16) ? 0 : 8;
  int v    = h >> 1;
  int k    = kb * 32 + ((v & 4) ? 16 : 0) + koff + ((v & 3) << 1) + (h & 1);
  int n    = ct * 16 + lh;
  out[t] = (_Float16)W[k * 128 + n];
}

// C[nrows x 128] = A[nrows x 128] * B[128 x 128] + bias, f16 in / f32 accum.
// One wave owns one 16-row tile and all 8 column tiles (8 accumulators);
// the A fragment is loaded once per k-step and reused across 8 WMMAs.
// Bsw is the pre-swizzled fragment-order weight matrix.
__global__ void gemm128_wmma_kernel(const _Float16* __restrict__ A,
                                    const _Float16* __restrict__ Bsw,
                                    const float* __restrict__ bias,
                                    float* __restrict__ C, int nrows) {
  const int lane = threadIdx.x & 31;
  const int wave = threadIdx.x >> 5;
  const int row0 = (blockIdx.x * 8 + wave) * 16;
  const int lh   = lane & 15;
  const int koff = (lane < 16) ? 0 : 8;

  int arow = row0 + lh;
  if (arow >= nrows) arow = (nrows > 0) ? (nrows - 1) : 0;  // clamp (guarded on store)

  v8f acc[8];
#pragma unroll
  for (int ct = 0; ct < 8; ++ct) acc[ct] = (v8f){};

#pragma unroll
  for (int kb = 0; kb < 4; ++kb) {
    // A 16x32 fragment: lane holds row (lane%16); halves 0..7 -> K=kb*32+koff+h,
    // halves 8..15 -> K=kb*32+16+koff+(h-8). Two contiguous 16B chunks.
    const _Float16* ap = A + (long)arow * 128 + kb * 32 + koff;
    v16h a;
#pragma unroll
    for (int h = 0; h < 8; ++h) { a[h] = ap[h]; a[h + 8] = ap[16 + h]; }
#pragma unroll
    for (int ct = 0; ct < 8; ++ct) {
      v16h b = *(const v16h*)(Bsw + (((kb * 8 + ct) * 32) + lane) * 16);
      acc[ct] = __builtin_amdgcn_wmma_f32_16x16x32_f16(false, a, false, b,
                                                       (short)0, acc[ct], false, false);
    }
  }
  // C/D layout: VGPR v -> M = v + (lane<16?0:8), N = lane%16
#pragma unroll
  for (int ct = 0; ct < 8; ++ct) {
    int c = ct * 16 + lh;
    float bv = bias[c];
#pragma unroll
    for (int v = 0; v < 8; ++v) {
      int r = row0 + v + ((lane < 16) ? 0 : 8);
      if (r < nrows) C[(long)r * 128 + c] = acc[ct][v] + bv;
    }
  }
}

// Layer 1: per (edge, head) logit = sum_c leakyrelu(xl[src]+xr[dst]) * att ; atomicMax into m.
__global__ void l1_logits_kernel(const float* __restrict__ xl, const float* __restrict__ xr,
                                 const int* __restrict__ ei, int E, int Ne,
                                 const float* __restrict__ att,
                                 float* __restrict__ logit, unsigned* __restrict__ mmax) {
  int t = blockIdx.x * blockDim.x + threadIdx.x;
  if (t >= Ne * 4) return;
  int e = t >> 2, h = t & 3;
  int src = (e < E) ? ei[e] : (e - E);
  int dst = (e < E) ? ei[E + e] : (e - E);
  const float4* pl = (const float4*)(xl + (long)src * 128 + h * 32);
  const float4* pr = (const float4*)(xr + (long)dst * 128 + h * 32);
  const float4* pa = (const float4*)(att + h * 32);
  float s = 0.f;
#pragma unroll
  for (int j = 0; j < 8; ++j) {
    float4 a = pl[j], b = pr[j], w = pa[j];
    float v;
    v = a.x + b.x; v = v > 0.f ? v : NEG_SLOPE * v; s += v * w.x;
    v = a.y + b.y; v = v > 0.f ? v : NEG_SLOPE * v; s += v * w.y;
    v = a.z + b.z; v = v > 0.f ? v : NEG_SLOPE * v; s += v * w.z;
    v = a.w + b.w; v = v > 0.f ? v : NEG_SLOPE * v; s += v * w.w;
  }
  logit[t] = s;
  atomicMax(&mmax[dst * 4 + h], fmap(s));
}

__global__ void l1_expsum_kernel(const int* __restrict__ ei, int E, int Ne,
                                 float* __restrict__ logit,
                                 const unsigned* __restrict__ mmax,
                                 float* __restrict__ denom) {
  int t = blockIdx.x * blockDim.x + threadIdx.x;
  if (t >= Ne * 4) return;
  int e = t >> 2, h = t & 3;
  int dst = (e < E) ? ei[E + e] : (e - E);
  float p = expf(logit[t] - funmap(mmax[dst * 4 + h]));
  logit[t] = p;
  atomicAdd(&denom[dst * 4 + h], p);
}

__global__ void l1_aggregate_kernel(const int* __restrict__ ei, int E, int Ne,
                                    const float* __restrict__ xl,
                                    const float* __restrict__ p,
                                    const float* __restrict__ denom,
                                    float* __restrict__ out) {
  int t = blockIdx.x * blockDim.x + threadIdx.x;       // Ne*128 < 2^31
  if (t >= Ne * 128) return;
  int e = t >> 7, c = t & 127, h = c >> 5;
  int src = (e < E) ? ei[e] : (e - E);
  int dst = (e < E) ? ei[E + e] : (e - E);
  float alpha = p[e * 4 + h] / denom[dst * 4 + h];
  atomicAdd(&out[(long)dst * 128 + c], xl[(long)src * 128 + c] * alpha);
}

// bias + ELU + f16 convert (input to layer-2 GEMMs)
__global__ void l1_finish_kernel(const float* __restrict__ acc, const float* __restrict__ bias,
                                 _Float16* __restrict__ hh, int n) {
  int i = blockIdx.x * blockDim.x + threadIdx.x;
  if (i < n) {
    float v = acc[i] + bias[i & 127];
    v = v > 0.f ? v : expm1f(v);
    hh[i] = (_Float16)v;
  }
}

// Layer 2: heads=1, C=128; one thread per edge.
__global__ void l2_logits_kernel(const float* __restrict__ xl, const float* __restrict__ xr,
                                 const int* __restrict__ ei, int E, int Ne,
                                 const float* __restrict__ att,
                                 float* __restrict__ logit, unsigned* __restrict__ mmax) {
  int e = blockIdx.x * blockDim.x + threadIdx.x;
  if (e >= Ne) return;
  int src = (e < E) ? ei[e] : (e - E);
  int dst = (e < E) ? ei[E + e] : (e - E);
  const float4* pl = (const float4*)(xl + (long)src * 128);
  const float4* pr = (const float4*)(xr + (long)dst * 128);
  const float4* pa = (const float4*)att;
  float s = 0.f;
#pragma unroll
  for (int j = 0; j < 32; ++j) {
    float4 a = pl[j], b = pr[j], w = pa[j];
    float v;
    v = a.x + b.x; v = v > 0.f ? v : NEG_SLOPE * v; s += v * w.x;
    v = a.y + b.y; v = v > 0.f ? v : NEG_SLOPE * v; s += v * w.y;
    v = a.z + b.z; v = v > 0.f ? v : NEG_SLOPE * v; s += v * w.z;
    v = a.w + b.w; v = v > 0.f ? v : NEG_SLOPE * v; s += v * w.w;
  }
  logit[e] = s;
  atomicMax(&mmax[dst], fmap(s));
}

__global__ void l2_expsum_kernel(const int* __restrict__ ei, int E, int Ne,
                                 float* __restrict__ logit,
                                 const unsigned* __restrict__ mmax,
                                 float* __restrict__ denom) {
  int e = blockIdx.x * blockDim.x + threadIdx.x;
  if (e >= Ne) return;
  int dst = (e < E) ? ei[E + e] : (e - E);
  float p = expf(logit[e] - funmap(mmax[dst]));
  logit[e] = p;
  atomicAdd(&denom[dst], p);
}

__global__ void l2_aggregate_kernel(const int* __restrict__ ei, int E, int Ne,
                                    const float* __restrict__ xl,
                                    const float* __restrict__ p,
                                    const float* __restrict__ denom,
                                    float* __restrict__ out) {
  int t = blockIdx.x * blockDim.x + threadIdx.x;
  if (t >= Ne * 128) return;
  int e = t >> 7, c = t & 127;
  int src = (e < E) ? ei[e] : (e - E);
  int dst = (e < E) ? ei[E + e] : (e - E);
  float alpha = p[e] / denom[dst];
  atomicAdd(&out[(long)dst * 128 + c], xl[(long)src * 128 + c] * alpha);
}

__global__ void bias_add_kernel(float* __restrict__ out, const float* __restrict__ bias, int n) {
  int i = blockIdx.x * blockDim.x + threadIdx.x;
  if (i < n) out[i] += bias[i & 127];
}

extern "C" void kernel_launch(void* const* d_in, const int* in_sizes, int n_in,
                              void* d_out, int out_size, void* d_ws, size_t ws_size,
                              hipStream_t stream) {
  const float* x     = (const float*)d_in[0];
  const int*   ei    = (const int*)d_in[1];
  const float* W1l   = (const float*)d_in[2];
  const float* b1l   = (const float*)d_in[3];
  const float* W1r   = (const float*)d_in[4];
  const float* b1r   = (const float*)d_in[5];
  const float* att1  = (const float*)d_in[6];
  const float* bias1 = (const float*)d_in[7];
  const float* W2l   = (const float*)d_in[8];
  const float* b2l   = (const float*)d_in[9];
  const float* W2r   = (const float*)d_in[10];
  const float* b2r   = (const float*)d_in[11];
  const float* att2  = (const float*)d_in[12];
  const float* bias2 = (const float*)d_in[13];

  const int N  = in_sizes[0] / 128;
  const int E  = in_sizes[1] / 2;
  const int Ne = E + N;

  // workspace carve-out (256B aligned chunks)
  char* wsp = (char*)d_ws;
  size_t off = 0;
  auto take = [&](size_t bytes) -> void* {
    void* p = wsp + off;
    off += (bytes + 255) & ~(size_t)255;
    return p;
  };
  _Float16* xh    = (_Float16*)take((size_t)N * 128 * 2);   // x (layer1) / h (layer2) in f16
  _Float16* w1lh  = (_Float16*)take(128 * 128 * 2);         // swizzled fragment order
  _Float16* w1rh  = (_Float16*)take(128 * 128 * 2);
  _Float16* w2lh  = (_Float16*)take(128 * 128 * 2);
  _Float16* w2rh  = (_Float16*)take(128 * 128 * 2);
  float*    xl    = (float*)take((size_t)N * 128 * 4);      // left transform (both layers)
  float*    xr    = (float*)take((size_t)N * 128 * 4);      // right transform (both layers)
  float*    pbuf  = (float*)take((size_t)Ne * 4 * 4);       // logits -> exp(p) (layer2 uses Ne)
  unsigned* mmax  = (unsigned*)take((size_t)N * 4 * 4);     // mapped segment max
  float*    denom = (float*)take((size_t)N * 4 * 4);
  float*    acc1  = (float*)take((size_t)N * 128 * 4);      // layer-1 aggregation

  const int B = 256;
  auto nb = [](long long n) { return (unsigned)((n + 255) / 256); };
  const long long nNF = (long long)N * 128;
  const long long nE4 = (long long)Ne * 4;
  const long long nEC = (long long)Ne * 128;
  const unsigned rowTiles   = (unsigned)((N + 15) / 16);
  const unsigned gemmBlocks = (rowTiles + 7) / 8;

  // ---- layer 1 ----
  f32_to_f16_kernel<<<nb(nNF), B, 0, stream>>>(x, xh, (int)nNF);
  wswz_kernel<<<64, B, 0, stream>>>(W1l, w1lh);
  wswz_kernel<<<64, B, 0, stream>>>(W1r, w1rh);
  wswz_kernel<<<64, B, 0, stream>>>(W2l, w2lh);
  wswz_kernel<<<64, B, 0, stream>>>(W2r, w2rh);

  gemm128_wmma_kernel<<<gemmBlocks, B, 0, stream>>>(xh, w1lh, b1l, xl, N);
  gemm128_wmma_kernel<<<gemmBlocks, B, 0, stream>>>(xh, w1rh, b1r, xr, N);

  fill_u32_kernel<<<nb((long long)N * 4), B, 0, stream>>>(mmax, 0u, (long long)N * 4);
  fill_u32_kernel<<<nb((long long)N * 4), B, 0, stream>>>((unsigned*)denom, 0u, (long long)N * 4);
  fill_u32_kernel<<<nb(nNF), B, 0, stream>>>((unsigned*)acc1, 0u, nNF);

  l1_logits_kernel<<<nb(nE4), B, 0, stream>>>(xl, xr, ei, E, Ne, att1, pbuf, mmax);
  l1_expsum_kernel<<<nb(nE4), B, 0, stream>>>(ei, E, Ne, pbuf, mmax, denom);
  l1_aggregate_kernel<<<nb(nEC), B, 0, stream>>>(ei, E, Ne, xl, pbuf, denom, acc1);
  l1_finish_kernel<<<nb(nNF), B, 0, stream>>>(acc1, bias1, xh, (int)nNF);

  // ---- layer 2 ----
  gemm128_wmma_kernel<<<gemmBlocks, B, 0, stream>>>(xh, w2lh, b2l, xl, N);
  gemm128_wmma_kernel<<<gemmBlocks, B, 0, stream>>>(xh, w2rh, b2r, xr, N);

  fill_u32_kernel<<<nb((long long)N), B, 0, stream>>>(mmax, 0u, (long long)N);
  fill_u32_kernel<<<nb((long long)N), B, 0, stream>>>((unsigned*)denom, 0u, (long long)N);
  fill_u32_kernel<<<nb(nNF), B, 0, stream>>>((unsigned*)d_out, 0u, nNF);

  l2_logits_kernel<<<nb((long long)Ne), B, 0, stream>>>(xl, xr, ei, E, Ne, att2, pbuf, mmax);
  l2_expsum_kernel<<<nb((long long)Ne), B, 0, stream>>>(ei, E, Ne, pbuf, mmax, denom);
  l2_aggregate_kernel<<<nb(nEC), B, 0, stream>>>(ei, E, Ne, xl, pbuf, denom, (float*)d_out);
  bias_add_kernel<<<nb(nNF), B, 0, stream>>>((float*)d_out, bias2, (int)nNF);

  (void)n_in; (void)out_size; (void)ws_size;
}
